// Layer_20521353740330
// MI455X (gfx1250) — matrix operand order repaired and checked by reading, not compiled, not run on hardware
//
#include <hip/hip_runtime.h>
#include <hip/hip_bf16.h>
#include <stdint.h>

#define B_   32
#define S_   2048
#define D_   1024
#define V_   50257
#define NS_  8
#define CHUNK_ (S_ / NS_)      // 256 tokens per split
#define TEMP_  0.7f
#define NEGM_  -10000.0f
#define NEGINF_ (-3.402823466e+38f)

typedef __attribute__((ext_vector_type(16))) __bf16 v16bf;
typedef __attribute__((ext_vector_type(8)))  __bf16 v8bf;
typedef __attribute__((ext_vector_type(8)))  float  v8f;

// ---------------- wave / block reductions (wave32, 256-thread blocks) -------
__device__ __forceinline__ float waveReduceSum(float v) {
#pragma unroll
  for (int off = 16; off > 0; off >>= 1) v += __shfl_xor(v, off, 32);
  return v;
}
__device__ __forceinline__ float waveReduceMax(float v) {
#pragma unroll
  for (int off = 16; off > 0; off >>= 1) v = fmaxf(v, __shfl_xor(v, off, 32));
  return v;
}
__device__ __forceinline__ float blockSum256(float v, float* s8) {
  v = waveReduceSum(v);
  const int wv = threadIdx.x >> 5, ln = threadIdx.x & 31;
  __syncthreads();                 // protect any prior use of s8
  if (ln == 0) s8[wv] = v;
  __syncthreads();
  float r = 0.f;
#pragma unroll
  for (int i = 0; i < 8; ++i) r += s8[i];
  return r;
}
__device__ __forceinline__ float blockMax256(float v, float* s8) {
  v = waveReduceMax(v);
  const int wv = threadIdx.x >> 5, ln = threadIdx.x & 31;
  __syncthreads();
  if (ln == 0) s8[wv] = v;
  __syncthreads();
  float r = s8[0];
#pragma unroll
  for (int i = 1; i < 8; ++i) r = fmaxf(r, s8[i]);
  return r;
}

// ---------------- Kernel 1: per-(row, S-chunk) softmax-weighted partial -----
// grid = B*NS, block = 256. One pass for scores, one L2-resident pass for the
// weighted sum. Outputs per-chunk (m, l, acc[D]) flash-softmax partials.
__global__ __launch_bounds__(256)
void pool_partial(const float* __restrict__ batch, const float* __restrict__ wp,
                  float* __restrict__ pm, float* __restrict__ pl,
                  float* __restrict__ pacc) {
  const int b    = blockIdx.x / NS_;
  const int sp   = blockIdx.x % NS_;
  const int t0   = sp * CHUNK_;
  const int tid  = threadIdx.x;
  const int wave = tid >> 5, lane = tid & 31;

  __shared__ float sc[CHUNK_];
  __shared__ float redm[8];
  __shared__ float redl[8];

  // each lane owns w_pool[j*32 + lane], j = 0..31  (covers all D=1024)
  float wreg[32];
#pragma unroll
  for (int j = 0; j < 32; ++j) wreg[j] = wp[j * 32 + lane];

  const float* cbase = batch + ((size_t)b * S_ + t0) * D_;

  // phase 1: each wave scores 32 tokens (coalesced b32 loads, wave reduce)
  for (int t = wave * 32; t < wave * 32 + 32; ++t) {
    const float* x = cbase + (size_t)t * D_;
    float s = 0.f;
#pragma unroll
    for (int j = 0; j < 32; ++j) s = fmaf(x[j * 32 + lane], wreg[j], s);
    s = waveReduceSum(s);
    if (lane == 0) sc[t] = s;
  }
  __syncthreads();

  // phase 2: chunk softmax stats (each thread owns exactly one score)
  const float sv = sc[tid];
  const float m  = blockMax256(sv, redm);
  const float ev = __expf(sv - m);
  const float l  = blockSum256(ev, redl);
  sc[tid] = ev;                    // own-slot write, no cross dependency
  __syncthreads();

  // phase 3: weighted sum; thread owns d = tid*4 .. tid*4+3 (float4, coalesced)
  float4 acc = make_float4(0.f, 0.f, 0.f, 0.f);
  for (int t = 0; t < CHUNK_; ++t) {
    const float w = sc[t];
    const float4 x = *(const float4*)(cbase + (size_t)t * D_ + tid * 4);
    acc.x = fmaf(w, x.x, acc.x);
    acc.y = fmaf(w, x.y, acc.y);
    acc.z = fmaf(w, x.z, acc.z);
    acc.w = fmaf(w, x.w, acc.w);
  }
  const size_t pidx = (size_t)blockIdx.x;     // == b*NS_ + sp
  if (tid == 0) { pm[pidx] = m; pl[pidx] = l; }
  *(float4*)(pacc + pidx * D_ + tid * 4) = acc;
}

// ---------------- Kernel 2: merge chunk partials, emit bf16 pooled ----------
__global__ __launch_bounds__(256)
void combine_pool(const float* __restrict__ pm, const float* __restrict__ pl,
                  const float* __restrict__ pacc, __bf16* __restrict__ pooled) {
  const int b = blockIdx.x, tid = threadIdx.x;
  float M = NEGINF_;
#pragma unroll
  for (int i = 0; i < NS_; ++i) M = fmaxf(M, pm[b * NS_ + i]);
  float w[NS_];
  float L = 0.f;
#pragma unroll
  for (int i = 0; i < NS_; ++i) {
    w[i] = __expf(pm[b * NS_ + i] - M);
    L = fmaf(pl[b * NS_ + i], w[i], L);
  }
  const float inv = 1.f / L;
#pragma unroll
  for (int j = 0; j < 4; ++j) {
    const int d = tid * 4 + j;
    float s = 0.f;
#pragma unroll
    for (int i = 0; i < NS_; ++i)
      s = fmaf(pacc[(size_t)(b * NS_ + i) * D_ + d], w[i], s);
    pooled[(size_t)b * D_ + d] = (__bf16)(s * inv);
  }
}

// ---------------- Kernel 3: logits = TEMP*(pooled @ K + bias), WMMA bf16 ----
// block = 128 (4 waves); wave (mt,nt) owns the 16x16 tile (rows mt*16..+15,
// cols blk*32 + nt*16..+15). K loop: 32 per v_wmma_f32_16x16x32_bf16.
__global__ __launch_bounds__(128)
void gemm_wmma(const __bf16* __restrict__ A, const float* __restrict__ Km,
               const float* __restrict__ bias, float* __restrict__ logits) {
  const int wave  = threadIdx.x >> 5;
  const int lane  = threadIdx.x & 31;
  const int mt    = wave & 1;
  const int nt    = wave >> 1;
  const int lhalf = lane & 15;
  const bool hi   = lane >= 16;
  const int col   = blockIdx.x * 32 + nt * 16 + lhalf;
  const bool cv   = col < V_;
  const int colc  = cv ? col : (V_ - 1);     // clamp: keep loads in-bounds

  // A fragment: lane holds row M = mt*16 + lhalf for both half-waves
  const __bf16* Arow = A + (size_t)(mt * 16 + lhalf) * D_;

  v8f c = {0.f, 0.f, 0.f, 0.f, 0.f, 0.f, 0.f, 0.f};

  for (int k = 0; k < D_; k += 32) {
    // ---- A frag (contiguous k order, same as B): lanes<16 -> K {k..k+15},
    //      lanes>=16 -> K {k+16..k+31}; two 16B contiguous loads.
    const int ka = k + (hi ? 16 : 0);
    const v8bf a0 = *(const v8bf*)(Arow + ka);
    const v8bf a1 = *(const v8bf*)(Arow + ka + 8);
    v16bf a;
#pragma unroll
    for (int i = 0; i < 8; ++i) { a[i] = a0[i]; a[8 + i] = a1[i]; }

    // ---- B frag (32x16): VGPR i holds K rows kb+2i, kb+2i+1 at column lhalf;
    //      kb = k for lanes 0-15, k+16 for lanes 16-31. fp32 loads -> bf16.
    const int kb = k + (hi ? 16 : 0);
    const float* bp = Km + (size_t)kb * V_ + colc;
    v16bf bb;
#pragma unroll
    for (int i = 0; i < 8; ++i) {
      bb[2 * i]     = (__bf16)bp[(size_t)(2 * i) * V_];
      bb[2 * i + 1] = (__bf16)bp[(size_t)(2 * i + 1) * V_];
    }
    if (k + 32 < D_) __builtin_prefetch(bp + (size_t)32 * V_, 0, 1);

    // D = A x B + C   (emits v_wmma_f32_16x16x32_bf16)
    c = __builtin_amdgcn_wmma_f32_16x16x32_bf16(
        /*neg_a=*/false, a, /*neg_b=*/false, bb,
        /*c_mod=*/(short)0, c, /*reuse_a=*/false, /*reuse_b=*/false);
  }

  if (cv) {
    const float bcol = bias[col];
#pragma unroll
    for (int r = 0; r < 8; ++r) {
      const int m = mt * 16 + (hi ? 8 : 0) + r;   // C/D layout: VGPR r -> M=r / r+8
      float val = TEMP_ * (c[r] + bcol);
      if (col == 0) val += NEGM_;
      logits[(size_t)m * V_ + col] = val;
    }
  }
}

// ---------------- Kernel 4: top-p (nucleus) sampling, sort-free -------------
// Per row: max M, partition Z, then binary-search a logit threshold x with
// sum_{l-M>=x} exp(l-M) >= top_p*Z (L2-resident passes), then Gumbel-argmax
// over the kept set. xthr < 0 so the top-1 token is always kept.
__global__ __launch_bounds__(256)
void sample_topp(const float* __restrict__ logits, const float* __restrict__ tpp,
                 int* __restrict__ out) {
  const int b = blockIdx.x, tid = threadIdx.x;
  const int lane = tid & 31, wave = tid >> 5;
  __shared__ float s8[8];
  __shared__ float sbv[8];
  __shared__ int   sbi[8];

  const float* row = logits + (size_t)b * V_;
  const float top_p = tpp[0];

  float m = NEGINF_;
  for (int v = tid; v < V_; v += 256) m = fmaxf(m, row[v]);
  m = blockMax256(m, s8);

  float z = 0.f;
  for (int v = tid; v < V_; v += 256) z += __expf(row[v] - m);
  z = blockSum256(z, s8);

  const float target = top_p * z;
  float lo = -35.f, hi = 0.f;
  for (int it = 0; it < 28; ++it) {
    const float mid = 0.5f * (lo + hi);
    float s = 0.f;
    for (int v = tid; v < V_; v += 256) {
      const float d = row[v] - m;
      s += (d >= mid) ? __expf(d) : 0.f;
    }
    s = blockSum256(s, s8);          // uniform across block
    if (s >= target) lo = mid; else hi = mid;
  }
  const float xthr = lo;             // always < 0 -> top-1 kept

  float bestv = NEGINF_;
  int   besti = 0;
  for (int v = tid; v < V_; v += 256) {
    const float d = row[v] - m;
    if (d >= xthr) {
      uint32_t h = (uint32_t)(b * 50261 + v);
      h *= 2654435761u; h ^= h >> 16;
      h *= 2246822519u; h ^= h >> 13;
      h *= 3266489917u; h ^= h >> 16;
      const float u = (float)(h >> 8) * (1.0f / 16777216.0f) + 1e-12f;
      const float g = -__logf(-__logf(u) + 1e-20f);   // Gumbel(0,1)
      const float key = d + g;                        // log-prob const per row
      if (key > bestv) { bestv = key; besti = v; }
    }
  }
#pragma unroll
  for (int off = 16; off > 0; off >>= 1) {
    const float ov = __shfl_xor(bestv, off, 32);
    const int   oi = __shfl_xor(besti, off, 32);
    if (ov > bestv) { bestv = ov; besti = oi; }
  }
  __syncthreads();
  if (lane == 0) { sbv[wave] = bestv; sbi[wave] = besti; }
  __syncthreads();
  if (tid == 0) {
    float bv = sbv[0]; int bi = sbi[0];
#pragma unroll
    for (int i = 1; i < 8; ++i)
      if (sbv[i] > bv) { bv = sbv[i]; bi = sbi[i]; }
    out[b] = bi;
  }
}

// ---------------- launcher --------------------------------------------------
extern "C" void kernel_launch(void* const* d_in, const int* in_sizes, int n_in,
                              void* d_out, int out_size, void* d_ws, size_t ws_size,
                              hipStream_t stream) {
  (void)in_sizes; (void)n_in; (void)out_size; (void)ws_size;
  const float* batch = (const float*)d_in[0];
  const float* wpool = (const float*)d_in[1];
  const float* kern  = (const float*)d_in[2];
  const float* bias  = (const float*)d_in[3];
  const float* top_p = (const float*)d_in[4];
  int* out = (int*)d_out;   // reference output dtype is int32 tokens

  char* ws = (char*)d_ws;
  size_t off = 0;
  auto alloc = [&](size_t bytes) -> void* {
    off = (off + 255) & ~(size_t)255;
    void* p = ws + off;
    off += bytes;
    return p;
  };
  float*  pm     = (float*) alloc((size_t)B_ * NS_ * sizeof(float));
  float*  pl     = (float*) alloc((size_t)B_ * NS_ * sizeof(float));
  float*  pacc   = (float*) alloc((size_t)B_ * NS_ * D_ * sizeof(float));
  __bf16* pooled = (__bf16*)alloc((size_t)B_ * D_ * sizeof(__bf16));
  float*  logits = (float*) alloc((size_t)B_ * V_ * sizeof(float));

  pool_partial<<<B_ * NS_, 256, 0, stream>>>(batch, wpool, pm, pl, pacc);
  combine_pool<<<B_, 256, 0, stream>>>(pm, pl, pacc, pooled);
  const int nblk = (V_ + 31) / 32;                  // 1571 column slabs
  gemm_wmma<<<nblk, 128, 0, stream>>>(pooled, kern, bias, logits);
  sample_topp<<<B_, 256, 0, stream>>>(logits, top_p, out);
}
